// MessageFunction_26465588478646
// MI455X (gfx1250) — matrix operand order repaired and checked by reading, not compiled, not run on hardware
//
#include <hip/hip_runtime.h>

// Problem constants from the reference
#define Bdim 16
#define Ndim 1024
#define Ddim 64
#define Edim 8

typedef __attribute__((ext_vector_type(16))) _Float16     v16h;
typedef __attribute__((ext_vector_type(8)))  float        v8f;
typedef __attribute__((ext_vector_type(4)))  unsigned int u32x4;
typedef __attribute__((ext_vector_type(8)))  int          i32x8;
typedef __attribute__((ext_vector_type(4)))  int          i32x4;

// ---------------------------------------------------------------------------
// Phase 1: hwT[e2][b][i][w] = sum_j W[e2][i][j] * h[b][w][j]   (f16, i-major)
// e2 in [0,8) -> matrix_in, e2 in [8,16) -> matrix_out. WMMA f16 -> f32.
// A tile = W rows (16 i x 32 j), B tile = h^T (32 j x 16 w), D = 16 i x 16 w.
// ---------------------------------------------------------------------------
__global__ __launch_bounds__(256) void ggnn_hw_kernel(
    const float* __restrict__ node, const float* __restrict__ Win,
    const float* __restrict__ Wout, _Float16* __restrict__ hwT) {
  const int lane = threadIdx.x & 31;
  const int wave = threadIdx.x >> 5;
  const int e2   = blockIdx.x >> 4;   // 0..15
  const int b    = blockIdx.x & 15;
  const float* W = (e2 < Edim) ? (Win + (size_t)e2 * Ddim * Ddim)
                               : (Wout + (size_t)(e2 - Edim) * Ddim * Ddim);
  _Float16* dst   = hwT + (size_t)e2 * (Bdim * Ddim * Ndim) + (size_t)b * (Ddim * Ndim);
  const float* hb = node + (size_t)b * Ndim * Ddim;
  const int lo = lane & 15;
  const int hi = lane >> 4;
  const int g  = hi * 8;

  // 4 i-tiles x 64 w-tiles = 256 tiles per (e2,b); 8 waves -> 32 tiles/wave
  for (int tt = wave; tt < 256; tt += 8) {
    const int i0 = (tt >> 6) * 16;
    const int w0 = (tt & 63) * 16;

    // A (16x32 f16): lane holds row i0+lo, K(h) = hi*16 + h
    const float* wr = W + (size_t)(i0 + lo) * Ddim;
    v16h A0, A1;
#pragma unroll
    for (int h = 0; h < 16; ++h) {
      const int j = h + (hi << 4);
      A0[h] = (_Float16)wr[j];
      A1[h] = (_Float16)wr[j + 32];
    }
    // B (32x16 f16): lane holds column w0+lo, K = hi*16 + h (contiguous j)
    const float* hr = hb + (size_t)(w0 + lo) * Ddim + hi * 16;
    v16h B0, B1;
#pragma unroll
    for (int h = 0; h < 16; ++h) {
      B0[h] = (_Float16)hr[h];
      B1[h] = (_Float16)hr[h + 32];
    }
    v8f acc = {};
    acc = __builtin_amdgcn_wmma_f32_16x16x32_f16(false, A0, false, B0, (short)0, acc, false, false);
    acc = __builtin_amdgcn_wmma_f32_16x16x32_f16(false, A1, false, B1, (short)0, acc, false, false);
    // C layout: lane -> (N = lo = w), VGPR r -> M = r + hi*8 (= i)
#pragma unroll
    for (int r = 0; r < 8; ++r) {
      dst[(size_t)(i0 + r + hi * 8) * Ndim + w0 + lo] = (_Float16)acc[r];
    }
  }
}

// ---------------------------------------------------------------------------
// TDM: 3D tensor load to LDS. dims: x = 32 w (stride 1), y = 64 i (stride
// 1024), z = 8 e (stride 16*64*1024). data_size = 2 bytes. LDS packed [e][i][w].
// ---------------------------------------------------------------------------
__device__ __forceinline__ void tdm_load_chunk(unsigned ldsOff, unsigned long long ga) {
  u32x4 g0;
  g0.x = 1u;                                                 // count=1, user mode
  g0.y = ldsOff;                                             // LDS byte address
  g0.z = (unsigned)ga;                                       // global addr [31:0]
  g0.w = ((unsigned)(ga >> 32) & 0x01ffffffu) | (2u << 30);  // addr [56:32] | type=2
  i32x8 g1;
  g1[0] = 0x00010000;               // wg_mask=0, data_size=1 (2 bytes)
  g1[1] = (int)(1024u << 16);       // tensor_dim0 = 1024 (bits 63:48)
  g1[2] = (int)(64u << 16);         // tensor_dim1 = 64   (bits 95:80)
  g1[3] = (int)(32u << 16);         // tile_dim0   = 32   (bits 127:112)
  g1[4] = (int)(64u | (8u << 16));  // tile_dim1=64, tile_dim2=8
  g1[5] = 1024;                     // tensor_dim0_stride = 1024
  g1[6] = 0;                        // strides mid bits
  g1[7] = 16;                       // tensor_dim1_stride = 16<<16 = 1048576
  i32x4 g2;
  g2[0] = 8;                        // tensor_dim2 = 8
  g2[1] = 0; g2[2] = 0; g2[3] = 0;
  i32x4 g3;
  g3[0] = 0; g3[1] = 0; g3[2] = 0; g3[3] = 0;
  i32x8 g4;                         // extra operand in clang-23 6-arg form; unused -> 0
  g4[0] = 0; g4[1] = 0; g4[2] = 0; g4[3] = 0;
  g4[4] = 0; g4[5] = 0; g4[6] = 0; g4[7] = 0;
  __builtin_amdgcn_tensor_load_to_lds(g0, g1, g2, g3, g4, 0);
}

// ---------------------------------------------------------------------------
// Phase 2: per (b, 64-row v-block) workgroup of 4 waves; each wave owns a
// 16-row v tile and accumulates a_in / a_out (16 x 64 each) over K = w.
// A = on-the-fly 0/1 mask tiles from adj (per edge class), B = hw chunk staged
// in LDS by the Tensor Data Mover with ping-pong double buffering so the DMA
// of chunk k+1 overlaps the 64 WMMAs of chunk k.
// ---------------------------------------------------------------------------
__global__ __launch_bounds__(128) void ggnn_msg_kernel(
    const int* __restrict__ adj, const _Float16* __restrict__ hwT,
    const float* __restrict__ bias, float* __restrict__ out) {
  __shared__ __align__(32) _Float16 sIN [2][Edim * Ddim * 32];   // 2 x 32 KB
  __shared__ __align__(32) _Float16 sOUT[2][Edim * Ddim * 32];   // 2 x 32 KB
  const int lane = threadIdx.x & 31;
  const int wave = threadIdx.x >> 5;
  const int b    = blockIdx.y;
  const int v0   = blockIdx.x * 64 + wave * 16;
  const int lo   = lane & 15;
  const int hi   = lane >> 4;
  const int g    = hi * 8;      // unused; A-operand K base for this half-wave is hi*16
  const int* adjb = adj + (size_t)b * Ndim * Ndim;
  const unsigned long long hwBase  = (unsigned long long)hwT;
  const unsigned long long outOffs = (unsigned long long)Edim * Bdim * Ddim * Ndim * 2u;
  const unsigned ldsIn [2] = {(unsigned)(unsigned long long)(void*)sIN[0],
                              (unsigned)(unsigned long long)(void*)sIN[1]};
  const unsigned ldsOut[2] = {(unsigned)(unsigned long long)(void*)sOUT[0],
                              (unsigned)(unsigned long long)(void*)sOUT[1]};

  v8f accI[4] = {};
  v8f accO[4] = {};

  // Prologue: kick off chunk 0 into buffer 0.
  if (wave == 0) {
    const unsigned long long ga0 = hwBase + (unsigned long long)(b * Ddim * Ndim * 2);
    tdm_load_chunk(ldsIn[0], ga0);
    tdm_load_chunk(ldsOut[0], ga0 + outOffs);
  }

  for (int k = 0; k < Ndim / 32; ++k) {
    const int cur = k & 1;
    const int w0  = k * 32;
    if (wave == 0) {
      __builtin_amdgcn_s_wait_tensorcnt(0);   // chunk k landed in buf[cur]
    }
    __syncthreads();  // chunk k visible to all; all waves done with buf[cur^1]
    if (wave == 0 && (k + 1) < Ndim / 32) {
      const unsigned long long ga =
          hwBase + (unsigned long long)((b * Ddim * Ndim + (w0 + 32)) * 2);
      tdm_load_chunk(ldsIn[cur ^ 1], ga);          // overlap DMA with compute
      tdm_load_chunk(ldsOut[cur ^ 1], ga + outOffs);
    }

    // ---- a_in: mask rows from adj[b, v0+lo, w0 + K(h)] (coalesced int4)
    {
      const int* row = adjb + (size_t)(v0 + lo) * Ndim + w0 + hi * 16;
      const int4 q0 = *(const int4*)(row + 0);
      const int4 q1 = *(const int4*)(row + 4);
      const int4 q2 = *(const int4*)(row + 8);
      const int4 q3 = *(const int4*)(row + 12);
      const int va[16] = {q0.x, q0.y, q0.z, q0.w, q1.x, q1.y, q1.z, q1.w,
                          q2.x, q2.y, q2.z, q2.w, q3.x, q3.y, q3.z, q3.w};
#pragma unroll
      for (int e = 0; e < Edim; ++e) {
        v16h m;
#pragma unroll
        for (int h = 0; h < 16; ++h)
          m[h] = (va[h] == e) ? (_Float16)1.0f : (_Float16)0.0f;
#pragma unroll
        for (int t = 0; t < 4; ++t) {
          const v16h bt =
              *(const v16h*)(sIN[cur] + ((e * Ddim + t * 16 + lo) * 32 + hi * 16));
          accI[t] = __builtin_amdgcn_wmma_f32_16x16x32_f16(
              false, m, false, bt, (short)0, accI[t], false, false);
        }
      }
    }
    // ---- a_out: transposed mask, adj[b, w0 + K(h), v0+lo]
    {
      int vb[16];
#pragma unroll
      for (int h = 0; h < 16; ++h) {
        const int K = h + hi * 16;
        vb[h] = adjb[(size_t)(w0 + K) * Ndim + v0 + lo];
      }
#pragma unroll
      for (int e = 0; e < Edim; ++e) {
        v16h m;
#pragma unroll
        for (int h = 0; h < 16; ++h)
          m[h] = (vb[h] == e) ? (_Float16)1.0f : (_Float16)0.0f;
#pragma unroll
        for (int t = 0; t < 4; ++t) {
          const v16h bt =
              *(const v16h*)(sOUT[cur] + ((e * Ddim + t * 16 + lo) * 32 + hi * 16));
          accO[t] = __builtin_amdgcn_wmma_f32_16x16x32_f16(
              false, m, false, bt, (short)0, accO[t], false, false);
        }
      }
    }
  }

  // Epilogue: out[b, v, 0:64] = a_in + bias[0:64]; out[b, v, 64:128] = a_out + bias[64:128]
#pragma unroll
  for (int t = 0; t < 4; ++t) {
    const float bi = bias[t * 16 + lo];
    const float bo = bias[Ddim + t * 16 + lo];
#pragma unroll
    for (int r = 0; r < 8; ++r) {
      float* o = out + (size_t)(b * Ndim + v0 + r + hi * 8) * (2 * Ddim);
      o[t * 16 + lo]        = accI[t][r] + bi;
      o[Ddim + t * 16 + lo] = accO[t][r] + bo;
    }
  }
}

// ---------------------------------------------------------------------------
extern "C" void kernel_launch(void* const* d_in, const int* in_sizes, int n_in,
                              void* d_out, int out_size, void* d_ws, size_t ws_size,
                              hipStream_t stream) {
  (void)in_sizes; (void)n_in; (void)out_size; (void)ws_size;
  const float* node = (const float*)d_in[0];   // [B,N,D]
  const int*   adjm = (const int*)d_in[1];     // [B,N,N]
  const float* Win  = (const float*)d_in[2];   // [E,D,D]
  const float* Wout = (const float*)d_in[3];   // [E,D,D]
  const float* bias = (const float*)d_in[4];   // [2D]
  float* out = (float*)d_out;                  // [B,N,2D]
  _Float16* hwT = (_Float16*)d_ws;             // 2*E*B*D*N f16 = 32 MiB scratch

  ggnn_hw_kernel<<<dim3(256), dim3(256), 0, stream>>>(node, Win, Wout, hwT);
  ggnn_msg_kernel<<<dim3(16, 16), dim3(128), 0, stream>>>(adjm, hwT, bias, out);
}
